// MultiHeadSelfAttention_34325378630073
// MI455X (gfx1250) — matrix operand order repaired and checked by reading, not compiled, not run on hardware
//
#include <hip/hip_runtime.h>

// ---------------------------------------------------------------------------
// MI455X (gfx1250) multi-head self-attention, bf16 WMMA + fp32 accumulate,
// async global->LDS double-buffered pipelines, DPP softmax, exp2 domain.
//   EMB=1024, HEADS=16, HEAD_DIM=64, B=2, N=2048  ->  M = B*N = 4096 tokens.
// ---------------------------------------------------------------------------

typedef unsigned short ushort_t;
typedef __attribute__((ext_vector_type(16))) __bf16 bf16x16;
typedef __attribute__((ext_vector_type(8)))  float  f32x8;
typedef __attribute__((ext_vector_type(4)))  int    i32x4;

union Frag16 { bf16x16 v; uint4 q[2]; };   // 32 bytes: one 16-bit A/B fragment

__device__ __forceinline__ ushort_t f2bf(float f) {
  unsigned u = __float_as_uint(f);
  u += 0x7fffu + ((u >> 16) & 1u);         // round-to-nearest-even
  return (ushort_t)(u >> 16);
}

// ---- CDNA5 async global->LDS copy (ASYNCcnt path), with sync fallback -----
#if defined(__has_builtin)
#if __has_builtin(__builtin_amdgcn_global_load_async_to_lds_b128) && \
    __has_builtin(__builtin_amdgcn_s_wait_asynccnt)
#define HAVE_ASYNC_LDS 1
#endif
#if __has_builtin(__builtin_amdgcn_mov_dpp8) && \
    __has_builtin(__builtin_amdgcn_update_dpp)
#define HAVE_DPP 1
#endif
#endif

#define AS1 __attribute__((address_space(1)))
#define AS3 __attribute__((address_space(3)))

__device__ __forceinline__ void cp16(const ushort_t* g, ushort_t* l) {
#if defined(HAVE_ASYNC_LDS)
  __builtin_amdgcn_global_load_async_to_lds_b128(
      (AS1 i32x4*)g, (AS3 i32x4*)l, 0, 0);
#else
  *(uint4*)l = *(const uint4*)g;
#endif
}

template <int N>
__device__ __forceinline__ void async_wait() {
#if defined(HAVE_ASYNC_LDS)
  __builtin_amdgcn_s_wait_asynccnt(N);
#endif
}

// ---- 16-lane-half butterfly reductions on the VALU via DPP ----------------
#define DPP8_SEL(a,b,c,d,e,f,g,h) \
  ((a)|((b)<<3)|((c)<<6)|((d)<<9)|((e)<<12)|((f)<<15)|((g)<<18)|((h)<<21))
#define DPP8_XOR1 DPP8_SEL(1,0,3,2,5,4,7,6)
#define DPP8_XOR2 DPP8_SEL(2,3,0,1,6,7,4,5)
#define DPP8_XOR4 DPP8_SEL(4,5,6,7,0,1,2,3)

#if defined(HAVE_DPP)
template <int SEL>
__device__ __forceinline__ float dpp8f(float x) {
  return __int_as_float(__builtin_amdgcn_mov_dpp8(__float_as_int(x), SEL));
}
__device__ __forceinline__ float rowmirrorf(float x) {   // lane ^ 15 in DPP row
  return __int_as_float(__builtin_amdgcn_update_dpp(
      0, __float_as_int(x), 0x140 /*ROW_MIRROR*/, 0xf, 0xf, true));
}
__device__ __forceinline__ float halfmax16(float v) {
  v = fmaxf(v, dpp8f<DPP8_XOR1>(v));
  v = fmaxf(v, dpp8f<DPP8_XOR2>(v));
  v = fmaxf(v, dpp8f<DPP8_XOR4>(v));
  v = fmaxf(v, rowmirrorf(v));   // uniform in 8-groups => ^15 acts as ^8
  return v;
}
__device__ __forceinline__ float halfsum16(float v) {
  v += dpp8f<DPP8_XOR1>(v);
  v += dpp8f<DPP8_XOR2>(v);
  v += dpp8f<DPP8_XOR4>(v);
  v += rowmirrorf(v);
  return v;
}
#else
__device__ __forceinline__ float halfmax16(float v) {
  v = fmaxf(v, __shfl_xor(v, 1, 32));
  v = fmaxf(v, __shfl_xor(v, 2, 32));
  v = fmaxf(v, __shfl_xor(v, 4, 32));
  v = fmaxf(v, __shfl_xor(v, 8, 32));
  return v;
}
__device__ __forceinline__ float halfsum16(float v) {
  v += __shfl_xor(v, 1, 32);
  v += __shfl_xor(v, 2, 32);
  v += __shfl_xor(v, 4, 32);
  v += __shfl_xor(v, 8, 32);
  return v;
}
#endif

// ------------------------------- fp32 -> bf16 ------------------------------
__global__ void cvt_bf16_kernel(const float* __restrict__ in,
                                ushort_t* __restrict__ out, int n) {
  int i = blockIdx.x * blockDim.x + threadIdx.x;
  if (i < n) out[i] = f2bf(in[i]);
}

// fp32 [1024 k][1024 n] -> bf16 transposed [1024 n][1024 k]
__global__ __launch_bounds__(256)
void cvt_transpose_kernel(const float* __restrict__ in,
                          ushort_t* __restrict__ out) {
  __shared__ float tile[32][33];
  const int tx = threadIdx.x & 31, ty = threadIdx.x >> 5;   // 32 x 8
  const int nb = blockIdx.x * 32, kb = blockIdx.y * 32;
#pragma unroll
  for (int i = 0; i < 32; i += 8)
    tile[ty + i][tx] = in[(size_t)(kb + ty + i) * 1024 + nb + tx];
  __syncthreads();
#pragma unroll
  for (int i = 0; i < 32; i += 8)
    out[(size_t)(nb + ty + i) * 1024 + kb + tx] = f2bf(tile[tx][ty + i]);
}

// ------------------------------- GEMM + bias -------------------------------
// C[4096,1024] = (A[4096,1024](bf16 rm) * Wt[1024 n][1024 k](bf16) + bias)*os
// OMODE: 0 = fp32 row-major, 1 = bf16 row-major, 2 = bf16 V^T per head
// Tile 128x128, 256 thr = 8 waves (4M x 2N), wave = 32x64 = 2x4 WMMA accums.
// Double-buffered LDS, async global->LDS prefetch of the next K-tile.
#define PADK 40   // 32 K elems padded to 40 (80B row stride, 16B multiple)
#define GK   1024
#define GN   1024

template <int OMODE>
__global__ __launch_bounds__(256)
void gemm_bf16_kernel(const ushort_t* __restrict__ A,
                      const ushort_t* __restrict__ Wt,
                      const float* __restrict__ bias,
                      float* __restrict__ outF, ushort_t* __restrict__ outB,
                      float oscale) {
  __shared__ ushort_t at[2][128][PADK];
  __shared__ ushort_t bt[2][128][PADK];   // [n][k]

  const int tid  = threadIdx.x;
  const int wave = tid >> 5;
  const int lane = tid & 31;
  const int half = lane >> 4;
  const int lr   = lane & 15;
  const int wm   = wave & 3;
  const int wn   = wave >> 2;
  const int bm   = blockIdx.x * 128;
  const int bn   = blockIdx.y * 128;

  auto fill = [&](int fb, int kt) {
    int r = tid >> 1, s = (tid & 1) * 16;
    cp16(A  + (size_t)(bm + r) * GK + kt + s,     &at[fb][r][s]);
    cp16(A  + (size_t)(bm + r) * GK + kt + s + 8, &at[fb][r][s + 8]);
    cp16(Wt + (size_t)(bn + r) * GK + kt + s,     &bt[fb][r][s]);
    cp16(Wt + (size_t)(bn + r) * GK + kt + s + 8, &bt[fb][r][s + 8]);
  };

  f32x8 acc[2][4] = {};

  fill(0, 0);
  for (int kt = 0; kt < GK; kt += 32) {
    const int buf = (kt >> 5) & 1;
    if (kt + 32 < GK) { fill(buf ^ 1, kt + 32); async_wait<4>(); }
    else              { async_wait<0>(); }
    __syncthreads();                       // tile[buf] ready for everyone

    Frag16 afr[2];
#pragma unroll
    for (int mt = 0; mt < 2; ++mt) {       // A layout: half picks K block of 8
      int row = wm * 32 + mt * 16 + lr;
      int lo  = half * 8;
      afr[mt].q[0] = *(const uint4*)&at[buf][row][lo];
      afr[mt].q[1] = *(const uint4*)&at[buf][row][lo + 16];
    }
    Frag16 bfr[4];
#pragma unroll
    for (int nt = 0; nt < 4; ++nt) {       // B layout: half picks K block of 8
      int col = wn * 64 + nt * 16 + lr;
      int ko  = half * 8;
      bfr[nt].q[0] = *(const uint4*)&bt[buf][col][ko];
      bfr[nt].q[1] = *(const uint4*)&bt[buf][col][ko + 16];
    }
#pragma unroll
    for (int mt = 0; mt < 2; ++mt)
#pragma unroll
      for (int nt = 0; nt < 4; ++nt)
        acc[mt][nt] = __builtin_amdgcn_wmma_f32_16x16x32_bf16(
            false, afr[mt].v, false, bfr[nt].v,
            (short)0, acc[mt][nt], false, false);
    __syncthreads();                       // done reading buf before refill
  }

  // epilogue: bias + scale + store (compile-time mode & strides)
#pragma unroll
  for (int mt = 0; mt < 2; ++mt)
#pragma unroll
    for (int nt = 0; nt < 4; ++nt) {
      int col = bn + wn * 64 + nt * 16 + lr;
      float bv = bias[col];
#pragma unroll
      for (int r = 0; r < 8; ++r) {
        int row = bm + wm * 32 + mt * 16 + r + half * 8;
        float v = (acc[mt][nt][r] + bv) * oscale;
        if (OMODE == 0) {
          outF[(size_t)row * GN + col] = v;
        } else if (OMODE == 1) {
          outB[(size_t)row * GN + col] = f2bf(v);
        } else {                           // V^T per head: [(b*16+h)*64+d][j]
          int bb = row >> 11, j = row & 2047;
          int h  = col >> 6,  d = col & 63;
          outB[((size_t)((bb * 16 + h) * 64 + d) << 11) + j] = f2bf(v);
        }
      }
    }
}

// ------------------------------ Flash attention ----------------------------
// Grid: (N/128 query blocks, B*HEADS). 256 thr = 8 waves, wave = 16 q rows.
// BK=64 keys/iteration: halves per-key softmax/rescale overhead vs BK=32.
// Q is pre-scaled by log2e/sqrt(64) -> all exponentials are raw exp2f.
// K tile [j][d] and V^T tile [d][j] are straight async copies, double-buffered.
#define ATT_N  2048
#define ATT_D  1024
#define HD     64
#define BK     64
#define PADD   72   // 64 elems padded to 72 (144B row stride, 16B multiple)

__global__ __launch_bounds__(256)
void flash_attn_kernel(const ushort_t* __restrict__ Qb,   // pre-scaled Q
                       const ushort_t* __restrict__ Kb,
                       const ushort_t* __restrict__ Vt,   // [bh][d][n] bf16
                       ushort_t* __restrict__ Ob) {
  __shared__ ushort_t kt_s[2][BK][PADD];     // K tile  [j][d]
  __shared__ ushort_t vt_s[2][HD][PADD];     // V^T tile [d][j]
  __shared__ ushort_t pb[8][16][PADD];       // per-wave P tile [row][j]

  const int tid  = threadIdx.x;
  const int wave = tid >> 5;
  const int lane = tid & 31;
  const int half = lane >> 4;
  const int lr   = lane & 15;

  const int bh = blockIdx.y;
  const int b  = bh >> 4;
  const int h  = bh & 15;
  const int qbase = blockIdx.x * 128;

  const ushort_t* Qh = Qb + (size_t)b * ATT_N * ATT_D + h * HD;
  const ushort_t* Kh = Kb + (size_t)b * ATT_N * ATT_D + h * HD;
  const ushort_t* Vh = Vt + ((size_t)bh * HD << 11);          // [d][2048]

  auto fill = [&](int fb, int jbase) {
    int j = tid >> 2, c = (tid & 3) * 16;      // K: 64 rows x 64 d
    cp16(Kh + (size_t)(jbase + j) * ATT_D + c,     &kt_s[fb][j][c]);
    cp16(Kh + (size_t)(jbase + j) * ATT_D + c + 8, &kt_s[fb][j][c + 8]);
    //                                         // V^T: 64 rows x 64 j
    cp16(Vh + ((size_t)j << 11) + jbase + c,       &vt_s[fb][j][c]);
    cp16(Vh + ((size_t)j << 11) + jbase + c + 8,   &vt_s[fb][j][c + 8]);
  };

  // resident Q fragments (16 rows x 64 d -> 2 k-steps of 32)
  Frag16 qf[2];
  {
    int row = qbase + wave * 16 + lr;
#pragma unroll
    for (int ks = 0; ks < 2; ++ks) {
      const ushort_t* p = Qh + (size_t)row * ATT_D + ks * 32 + half * 8;
      qf[ks].q[0] = *(const uint4*)p;
      qf[ks].q[1] = *(const uint4*)(p + 16);
    }
  }

  f32x8 o[4] = {};
  float m[8], l[8];
#pragma unroll
  for (int r = 0; r < 8; ++r) { m[r] = -1e30f; l[r] = 0.0f; }

  fill(0, 0);
  for (int jb = 0; jb < ATT_N; jb += BK) {
    const int buf = (jb >> 6) & 1;
    if (jb + BK < ATT_N) { fill(buf ^ 1, jb + BK); async_wait<4>(); }
    else                 { async_wait<0>(); }
    __syncthreads();

    // --- S = Q * K^T (16 x 64 per wave; log2e/sqrt(hd) pre-folded into Q) ---
    f32x8 s[4] = {};
#pragma unroll
    for (int jt = 0; jt < 4; ++jt) {
#pragma unroll
      for (int ks = 0; ks < 2; ++ks) {
        Frag16 kf;
        int j = jt * 16 + lr;
        kf.q[0] = *(const uint4*)&kt_s[buf][j][ks * 32 + half * 8];
        kf.q[1] = *(const uint4*)&kt_s[buf][j][ks * 32 + half * 8 + 16];
        s[jt] = __builtin_amdgcn_wmma_f32_16x16x32_bf16(
            false, qf[ks].v, false, kf.v, (short)0, s[jt], false, false);
      }
    }

    // --- online softmax in exp2 domain (row = one 16-lane half; DPP) ---
    float alpha[8];
#pragma unroll
    for (int r = 0; r < 8; ++r) {
      float mx = fmaxf(fmaxf(s[0][r], s[1][r]), fmaxf(s[2][r], s[3][r]));
      mx = halfmax16(mx);
      float mn = fmaxf(m[r], mx);
      alpha[r] = exp2f(m[r] - mn);
      float p0 = exp2f(s[0][r] - mn);
      float p1 = exp2f(s[1][r] - mn);
      float p2 = exp2f(s[2][r] - mn);
      float p3 = exp2f(s[3][r] - mn);
      float rs = halfsum16((p0 + p1) + (p2 + p3));
      l[r] = l[r] * alpha[r] + rs;
      m[r] = mn;
      s[0][r] = p0; s[1][r] = p1; s[2][r] = p2; s[3][r] = p3;
    }

    // --- P: C-layout -> LDS -> A-layout (bf16), wave-private buffer ---
#pragma unroll
    for (int jt = 0; jt < 4; ++jt)
#pragma unroll
      for (int r = 0; r < 8; ++r)
        pb[wave][r + half * 8][jt * 16 + lr] = f2bf(s[jt][r]);

#pragma unroll
    for (int dt = 0; dt < 4; ++dt)
#pragma unroll
      for (int r = 0; r < 8; ++r) o[dt][r] *= alpha[r];

    Frag16 pf[2];
#pragma unroll
    for (int ks = 0; ks < 2; ++ks) {
      pf[ks].q[0] = *(const uint4*)&pb[wave][lr][ks * 32 + half * 8];
      pf[ks].q[1] = *(const uint4*)&pb[wave][lr][ks * 32 + half * 8 + 16];
    }

    // --- O += P * V (inner dim 64 keys = 2 k-steps) ---
#pragma unroll
    for (int dt = 0; dt < 4; ++dt) {
#pragma unroll
      for (int ks = 0; ks < 2; ++ks) {
        Frag16 vf;
        int d = dt * 16 + lr;
        vf.q[0] = *(const uint4*)&vt_s[buf][d][ks * 32 + half * 8];
        vf.q[1] = *(const uint4*)&vt_s[buf][d][ks * 32 + half * 8 + 16];
        o[dt] = __builtin_amdgcn_wmma_f32_16x16x32_bf16(
            false, pf[ks].v, false, vf.v, (short)0, o[dt], false, false);
      }
    }
    __syncthreads();                       // done reading buf before refill
  }

  // finalize: O /= l, write bf16 head output [token][h*64+d]
#pragma unroll
  for (int dt = 0; dt < 4; ++dt)
#pragma unroll
    for (int r = 0; r < 8; ++r) {
      float inv = 1.0f / l[r];
      int row = qbase + wave * 16 + r + half * 8;
      int col = h * HD + dt * 16 + lr;
      Ob[((size_t)b * ATT_N + row) * ATT_D + col] = f2bf(o[dt][r] * inv);
    }
}

// ------------------------------- launcher ----------------------------------
extern "C" void kernel_launch(void* const* d_in, const int* in_sizes, int n_in,
                              void* d_out, int out_size, void* d_ws, size_t ws_size,
                              hipStream_t stream) {
  const float* x  = (const float*)d_in[0];
  const float* wq = (const float*)d_in[1];
  const float* bq = (const float*)d_in[2];
  const float* wk = (const float*)d_in[3];
  const float* bk = (const float*)d_in[4];
  const float* wv = (const float*)d_in[5];
  const float* bv = (const float*)d_in[6];
  const float* wo = (const float*)d_in[7];
  const float* bo = (const float*)d_in[8];

  const int M = 4096, D = 1024;
  const size_t xe = (size_t)M * D;           // 4,194,304 elems
  const size_t we = (size_t)D * D;           // 1,048,576 elems

  ushort_t* xb  = (ushort_t*)d_ws;           // x, bf16
  ushort_t* wqt = xb  + xe;                  // W^T bf16 (n-major)
  ushort_t* wkt = wqt + we;
  ushort_t* wvt = wkt + we;
  ushort_t* wot = wvt + we;
  ushort_t* Qbf = wot + we;                  // Q bf16 row-major (pre-scaled)
  ushort_t* Kbf = Qbf + xe;                  // K bf16 row-major
  ushort_t* Vtb = Kbf + xe;                  // V bf16 transposed per head
  ushort_t* Abf = Vtb + xe;                  // attention output bf16

  // 1) conversions (weights transposed so GEMM B-tiles are straight copies)
  cvt_bf16_kernel<<<(int)((xe + 255) / 256), 256, 0, stream>>>(x, xb, (int)xe);
  dim3 tgrid(32, 32);
  cvt_transpose_kernel<<<tgrid, 256, 0, stream>>>(wq, wqt);
  cvt_transpose_kernel<<<tgrid, 256, 0, stream>>>(wk, wkt);
  cvt_transpose_kernel<<<tgrid, 256, 0, stream>>>(wv, wvt);
  cvt_transpose_kernel<<<tgrid, 256, 0, stream>>>(wo, wot);

  // 2) projections. Q pre-scale = log2(e)/sqrt(head_dim) so the softmax runs
  //    in the exp2 domain (saves the log2e mul inside every exponential).
  const float qscale = 0.125f * 1.44269504088896340736f;
  dim3 ggrid(M / 128, D / 128);
  gemm_bf16_kernel<1><<<ggrid, 256, 0, stream>>>(xb, wqt, bq, nullptr, Qbf, qscale);
  gemm_bf16_kernel<1><<<ggrid, 256, 0, stream>>>(xb, wkt, bk, nullptr, Kbf, 1.0f);
  gemm_bf16_kernel<2><<<ggrid, 256, 0, stream>>>(xb, wvt, bv, nullptr, Vtb, 1.0f);

  // 3) flash attention
  flash_attn_kernel<<<dim3(ATT_N / 128, 32), 256, 0, stream>>>(Qbf, Kbf, Vtb, Abf);

  // 4) output projection -> fp32 d_out
  gemm_bf16_kernel<0><<<ggrid, 256, 0, stream>>>(Abf, wot, bo, (float*)d_out, nullptr, 1.0f);
}
